// Morpher_74259984548380
// MI455X (gfx1250) — matrix operand order repaired and checked by reading, not compiled, not run on hardware
//
#include <hip/hip_runtime.h>

// ---------------- problem constants ----------------
constexpr int IO_    = 1024;
constexpr int DSLOT  = 128;
constexpr int N_     = 4;
constexpr int K_     = 3;
constexpr int S_     = 7;
constexpr int D_     = DSLOT * S_;     // 896
constexpr int RANK_  = 256;
constexpr int HID_   = 3584;
constexpr int Bb_    = 4;
constexpr int T_     = 1024;
constexpr int M_     = Bb_ * T_;       // 4096 token rows
constexpr int ND_    = N_ * D_;        // 3584
constexpr int MN_    = M_ * N_;        // 16384
constexpr float EPS_ = 1e-5f;

// ACTIVE_SLOTS[phase][k], HEAD_ASSIGN[n][s] (precomputed from SCALES=[1,2,4])
__device__ __constant__ int c_active[4][3] = {{0,1,3},{0,2,4},{0,1,5},{0,2,6}};
__device__ __constant__ int c_head[4][7] = {
  {0, 4,6, 8,9,10,11},
  {1, 5,7, 9,10,11,8},
  {2, 6,4, 10,11,8,9},
  {3, 7,5, 11,8,9,10}};

// ---------------- vector types ----------------
typedef __attribute__((ext_vector_type(16))) __bf16 v16bf;
typedef __attribute__((ext_vector_type(8)))  __bf16 v8bf;
typedef __attribute__((ext_vector_type(8)))  float  v8f;
typedef int v4i_ __attribute__((vector_size(16)));   // matches async-LDS builtin pointee

#define LDSS 48   // padded LDS row stride (elements): 96B rows, 16B-aligned chunks

#define AS1 __attribute__((address_space(1)))
#define AS3 __attribute__((address_space(3)))

#if __has_builtin(__builtin_amdgcn_global_load_async_to_lds_b128)
#define HAVE_ASYNC_LDS 1
#endif

// copy 32 bytes global->LDS (two b128), async if available
__device__ __forceinline__ void stage32(const __bf16* g, __bf16* l) {
#ifdef HAVE_ASYNC_LDS
  __builtin_amdgcn_global_load_async_to_lds_b128((AS1 v4i_*)g,       (AS3 v4i_*)l,       0, 0);
  __builtin_amdgcn_global_load_async_to_lds_b128((AS1 v4i_*)(g + 8), (AS3 v4i_*)(l + 8), 0, 0);
#else
  *(v8bf*)(l)     = *(const v8bf*)(g);
  *(v8bf*)(l + 8) = *(const v8bf*)(g + 8);
#endif
}

__device__ __forceinline__ void stage_wait() {
#ifdef HAVE_ASYNC_LDS
#if __has_builtin(__builtin_amdgcn_s_wait_asynccnt)
  __builtin_amdgcn_s_wait_asynccnt(0);
#else
  asm volatile("s_wait_asynccnt 0x0" ::: "memory");
#endif
#endif
}

// A fragment (16x32 bf16): lane row = lane&15, k-half = (lane>>4)*8
__device__ __forceinline__ v16bf frag_a(const __bf16* s, int rowoff, int lane) {
  const int row = rowoff + (lane & 15);
  const int kh  = (lane >> 4) * 8;
  const __bf16* p = s + row * LDSS;
  v8bf lo = *(const v8bf*)(p + kh);
  v8bf hi = *(const v8bf*)(p + 16 + kh);
  v16bf r;
#pragma unroll
  for (int i = 0; i < 8; ++i) { r[i] = lo[i]; r[i + 8] = hi[i]; }
  return r;
}

// B fragment (32x16 bf16): lane col = lane&15, k-half = (lane>>4)*8
// LDS B tile is stored as [ncol][k] (i.e. already B^T)
__device__ __forceinline__ v16bf frag_b(const __bf16* s, int coloff, int lane) {
  const int col = coloff + (lane & 15);
  const int kb  = (lane >> 4) * 8;
  const __bf16* p = s + col * LDSS + kb;
  v8bf lo = *(const v8bf*)(p);
  v8bf hi = *(const v8bf*)(p + 16);
  v16bf r;
#pragma unroll
  for (int i = 0; i < 8; ++i) { r[i] = lo[i]; r[i + 8] = hi[i]; }
  return r;
}

// Core: C[m0:m0+128, n0:n0+128] = A(MxKd, row-major, lda) * B
//   BKN=false: B stored (Ncols x Kd) row-major (weight (out,in)) -> contiguous stage
//   BKN=true : B stored (Kd x Ncols) row-major (e.g. V matrix)   -> strided stage
// 256 threads = 8 waves; wave (w&3,w>>2) owns a 32x64 strip: 2 A-frags x 4 B-frags.
template <bool BKN, class F>
__device__ __forceinline__ void gemm_core(const __bf16* __restrict__ A, int lda,
                                          const __bf16* __restrict__ Bm, int ldb,
                                          int Kd, int m0, int n0, F epi) {
  __shared__ __bf16 sA[128 * LDSS];
  __shared__ __bf16 sB[128 * LDSS];
  const int tid  = threadIdx.x;
  const int wave = tid >> 5;
  const int lane = tid & 31;
  const int wr   = (wave & 3) * 32;   // wave row offset in tile
  const int wc   = (wave >> 2) * 64;  // wave col offset in tile
  const int lr   = tid & 127;         // staging row/col
  const int lh   = tid >> 7;          // staging 16-element half

  v8f acc[2][4];
#pragma unroll
  for (int s = 0; s < 2; ++s)
#pragma unroll
    for (int f = 0; f < 4; ++f)
#pragma unroll
      for (int j = 0; j < 8; ++j) acc[s][f][j] = 0.0f;

  for (int k0 = 0; k0 < Kd; k0 += 32) {
    __syncthreads();
    // stage A tile (128 rows x 32 k)
    stage32(A + (long)(m0 + lr) * lda + k0 + lh * 16, sA + lr * LDSS + lh * 16);
    if (BKN) { // stage B tile transposed: sB[n][k] = B[k][n]
#pragma unroll
      for (int j = 0; j < 16; ++j)
        sB[lr * LDSS + lh * 16 + j] =
            Bm[(long)(k0 + lh * 16 + j) * ldb + (n0 + lr)];
    } else {   // B already (n x k) row-major
      stage32(Bm + (long)(n0 + lr) * ldb + k0 + lh * 16, sB + lr * LDSS + lh * 16);
    }
    stage_wait();
    __syncthreads();

    v16bf a0 = frag_a(sA, wr, lane);
    v16bf a1 = frag_a(sA, wr + 16, lane);
#pragma unroll
    for (int f = 0; f < 4; ++f) {
      v16bf bf_ = frag_b(sB, wc + f * 16, lane);
      acc[0][f] = __builtin_amdgcn_wmma_f32_16x16x32_bf16(
          false, a0, false, bf_, (short)0, acc[0][f], false, false);
      acc[1][f] = __builtin_amdgcn_wmma_f32_16x16x32_bf16(
          false, a1, false, bf_, (short)0, acc[1][f], false, false);
    }
  }

  const int cl = lane & 15;
#pragma unroll
  for (int s = 0; s < 2; ++s) {
    const int rbase = m0 + wr + s * 16 + (lane >> 4) * 8;
#pragma unroll
    for (int f = 0; f < 4; ++f)
#pragma unroll
      for (int j = 0; j < 8; ++j)
        epi(rbase + j, n0 + wc + f * 16 + cl, acc[s][f][j]);
  }
}

// ---------------- elementwise conversions ----------------
__global__ __launch_bounds__(256) void k_cvt(const float* __restrict__ in,
                                             __bf16* __restrict__ out, long n) {
  long i = (long)blockIdx.x * 256 + threadIdx.x;
  if (i < n) out[i] = (__bf16)in[i];
}

// out[c*rows + r] = in[r*cols + c], batched along z
__global__ __launch_bounds__(256) void k_tcvt(const float* __restrict__ in,
                                              __bf16* __restrict__ out,
                                              int rows, int cols) {
  long boff = (long)blockIdx.z * rows * cols;
  int idx = blockIdx.x * 256 + threadIdx.x;
  if (idx < rows * cols) {
    int r = idx / cols, c = idx % cols;
    out[boff + (long)c * rows + r] = (__bf16)in[boff + idx];
  }
}

// ---------------- layernorm (block per row) ----------------
__global__ __launch_bounds__(256) void k_ln(const float* __restrict__ x,
                                            const float* __restrict__ g,
                                            const float* __restrict__ b,
                                            __bf16* __restrict__ out, int width) {
  __shared__ float sh[256], sh2[256];
  const long row = blockIdx.x;
  const float* xr = x + row * width;
  float s = 0.f, s2 = 0.f;
  for (int i = threadIdx.x; i < width; i += 256) {
    float v = xr[i]; s += v; s2 += v * v;
  }
  sh[threadIdx.x] = s; sh2[threadIdx.x] = s2; __syncthreads();
  for (int st = 128; st > 0; st >>= 1) {
    if (threadIdx.x < st) { sh[threadIdx.x] += sh[threadIdx.x + st];
                            sh2[threadIdx.x] += sh2[threadIdx.x + st]; }
    __syncthreads();
  }
  float mean = sh[0] / width;
  float var  = sh2[0] / width - mean * mean;
  float inv  = rsqrtf(var + EPS_);
  for (int i = threadIdx.x; i < width; i += 256) {
    float v = (xr[i] - mean) * inv * g[i] + b[i];
    out[row * width + i] = (__bf16)v;
  }
}

// ---------------- causal softmax, in place, bf16 rows of 1024 ----------------
__global__ __launch_bounds__(256) void k_softmax(__bf16* __restrict__ S) {
  __shared__ float sh[256];
  const long rowg = blockIdx.x;
  const int r = (int)(rowg & (T_ - 1));
  __bf16* p = S + rowg * T_;
  const int valid = r + 1;
  float mx = -3e38f;
  for (int i = threadIdx.x; i < valid; i += 256) mx = fmaxf(mx, (float)p[i]);
  sh[threadIdx.x] = mx; __syncthreads();
  for (int st = 128; st > 0; st >>= 1) {
    if (threadIdx.x < st) sh[threadIdx.x] = fmaxf(sh[threadIdx.x], sh[threadIdx.x + st]);
    __syncthreads();
  }
  mx = sh[0]; __syncthreads();
  float s = 0.f;
  for (int i = threadIdx.x; i < valid; i += 256) s += __expf((float)p[i] - mx);
  sh[threadIdx.x] = s; __syncthreads();
  for (int st = 128; st > 0; st >>= 1) {
    if (threadIdx.x < st) sh[threadIdx.x] += sh[threadIdx.x + st];
    __syncthreads();
  }
  float inv = 1.f / sh[0];
  for (int i = threadIdx.x; i < T_; i += 256) {
    float v = (i < valid) ? __expf((float)p[i] - mx) * inv : 0.f;
    p[i] = (__bf16)v;
  }
}

// ---------------- GEMM kernels (epilogue variants) ----------------
__global__ __launch_bounds__(256) void k_gemm_base(const __bf16* A, const __bf16* W,
                                                   float* out) {
  gemm_core<false>(A, IO_, W, IO_, IO_, blockIdx.y * 128, blockIdx.x * 128,
    [&](int r, int c, float v) { out[(long)r * D_ + c] = v; });
}

__global__ __launch_bounds__(256) void k_gemm_h(const __bf16* A, const __bf16* W,
                                                __bf16* out) {
  gemm_core<false>(A, IO_, W, IO_, IO_, blockIdx.y * 128, blockIdx.x * 128,
    [&](int r, int c, float v) { out[(long)r * RANK_ + c] = (__bf16)v; });
}

__global__ __launch_bounds__(256) void k_gemm_z(const __bf16* A, const __bf16* W,
                                                const float* __restrict__ basef,
                                                const float* __restrict__ beta_p,
                                                float* __restrict__ z) {
  float beta = *beta_p;
  gemm_core<false>(A, RANK_, W, RANK_, RANK_, blockIdx.y * 128, blockIdx.x * 128,
    [&](int r, int c, float v) {
      int d = c - (c / D_) * D_;
      z[(long)r * ND_ + c] = basef[(long)r * D_ + d] + beta * v;
    });
}

__global__ __launch_bounds__(256) void k_gemm_w1(const __bf16* A, const __bf16* W,
                                                 const float* __restrict__ b1,
                                                 __bf16* __restrict__ out) {
  gemm_core<false>(A, D_, W, D_, D_, blockIdx.y * 128, blockIdx.x * 128,
    [&](int r, int c, float v) {
      float u = v + b1[c];
      float ge = 0.5f * u * (1.f + erff(u * 0.7071067811865475f));
      out[(long)r * HID_ + c] = (__bf16)ge;
    });
}

__global__ __launch_bounds__(256) void k_gemm_w2(const __bf16* A, const __bf16* W,
                                                 const float* __restrict__ b2,
                                                 float* __restrict__ zh) {
  gemm_core<false>(A, HID_, W, HID_, HID_, blockIdx.y * 128, blockIdx.x * 128,
    [&](int r, int c, float v) {
      long idx = (long)r * D_ + c;
      zh[idx] = zh[idx] + v + b2[c];     // z_head = z + m (in place over z)
    });
}

__global__ __launch_bounds__(256) void k_gemm_qkv(const __bf16* __restrict__ hin,
                                                  const __bf16* __restrict__ wq,
                                                  const __bf16* __restrict__ wk,
                                                  const __bf16* __restrict__ wv,
                                                  const int* __restrict__ t_p,
                                                  __bf16* __restrict__ qb,
                                                  __bf16* __restrict__ kb,
                                                  __bf16* __restrict__ vb) {
  const int z = blockIdx.z, sel = z / 12, rem = z % 12, n = rem / 3, kk = rem % 3;
  const int t = *t_p, phase = ((t % N_) + N_) % N_;
  const int head = c_head[n][c_active[phase][kk]];
  const __bf16* W = (sel == 0 ? wq : sel == 1 ? wk : wv) + (long)head * DSLOT * D_;
  __bf16* out = (sel == 0 ? qb : sel == 1 ? kb : vb);
  const float sc = (sel == 0) ? 0.08838834764831845f : 1.0f;  // 1/sqrt(128) on q
  const int cb0 = (n * 3 + kk) * DSLOT;
  gemm_core<false>(hin + n * D_, ND_, W, D_, D_, blockIdx.y * 128, blockIdx.x * 128,
    [&](int r, int c, float v) {
      out[(long)r * (K_ * N_ * DSLOT) + cb0 + c] = (__bf16)(v * sc);
    });
}

__global__ __launch_bounds__(256) void k_gemm_scores(const __bf16* __restrict__ qb,
                                                     const __bf16* __restrict__ kb,
                                                     __bf16* __restrict__ S) {
  const int z = blockIdx.z, b = z / 12, n = (z % 12) / 3, kk = z % 3;
  const int m0 = blockIdx.y * 128, n0 = blockIdx.x * 128;
  if (n0 > m0 + 127) return;  // fully masked tile
  const int ldqk = K_ * N_ * DSLOT;  // 1536
  const long off = (long)b * T_ * ldqk + (n * 3 + kk) * DSLOT;
  __bf16* Sz = S + (long)z * T_ * T_;
  gemm_core<false>(qb + off, ldqk, kb + off, ldqk, DSLOT, m0, n0,
    [&](int r, int c, float v) {
      Sz[(long)r * T_ + c] = (__bf16)(c <= r ? v : -1e30f);
    });
}

__global__ __launch_bounds__(256) void k_gemm_attout(const __bf16* __restrict__ S,
                                                     const __bf16* __restrict__ vb,
                                                     const int* __restrict__ t_p,
                                                     float* __restrict__ zh) {
  const int z = blockIdx.z, b = z / 12, n = (z % 12) / 3, kk = z % 3;
  const int t = *t_p, phase = ((t % N_) + N_) % N_;
  const int slot = c_active[phase][kk];
  const int ldv = K_ * N_ * DSLOT;
  const __bf16* A  = S + (long)z * T_ * T_;
  const __bf16* Bm = vb + (long)b * T_ * ldv + (n * 3 + kk) * DSLOT;
  float* out = zh + (long)b * T_ * ND_ + n * D_ + slot * DSLOT;
  gemm_core<true>(A, T_, Bm, ldv, T_, blockIdx.y * 128, blockIdx.x * 128,
    [&](int r, int c, float v) {
      long idx = (long)r * ND_ + c;
      out[idx] += v;
    });
}

__global__ __launch_bounds__(256) void k_gemm_down(const __bf16* A, const __bf16* W,
                                                   __bf16* out) {
  gemm_core<false>(A, ND_, W, ND_, ND_, blockIdx.y * 128, blockIdx.x * 128,
    [&](int r, int c, float v) { out[(long)r * RANK_ + c] = (__bf16)v; });
}

__global__ __launch_bounds__(256) void k_gemm_up(const __bf16* A, const __bf16* W,
                                                 float* out) {
  gemm_core<false>(A, RANK_, W, RANK_, RANK_, blockIdx.y * 128, blockIdx.x * 128,
    [&](int r, int c, float v) { out[(long)r * IO_ + c] = v; });
}

// ---------------- host: workspace carve + launch sequence ----------------
extern "C" void kernel_launch(void* const* d_in, const int* in_sizes, int n_in,
                              void* d_out, int out_size, void* d_ws, size_t ws_size,
                              hipStream_t stream) {
  (void)in_sizes; (void)n_in; (void)out_size; (void)ws_size;
  const float* x        = (const float*)d_in[0];
  const int*   t_p      = (const int*)  d_in[1];
  const float* encbasew = (const float*)d_in[2];
  const float* encAw    = (const float*)d_in[3];
  const float* encB     = (const float*)d_in[4];
  const float* beta_p   = (const float*)d_in[5];
  const float* lnm_g    = (const float*)d_in[6];
  const float* lnm_b    = (const float*)d_in[7];
  const float* w1       = (const float*)d_in[8];
  const float* b1       = (const float*)d_in[9];
  const float* w2       = (const float*)d_in[10];
  const float* b2       = (const float*)d_in[11];
  const float* lna_g    = (const float*)d_in[12];
  const float* lna_b    = (const float*)d_in[13];
  const float* Wq       = (const float*)d_in[14];
  const float* Wk       = (const float*)d_in[15];
  const float* Wv       = (const float*)d_in[16];
  const float* dln_g    = (const float*)d_in[17];
  const float* dln_b    = (const float*)d_in[18];
  const float* wdown    = (const float*)d_in[19];
  const float* wup      = (const float*)d_in[20];
  float* y = (float*)d_out;

  char* w = (char*)d_ws;
  auto alloc = [&](size_t bytes) {
    char* p = w; w += (bytes + 255) & ~(size_t)255; return p;
  };
  __bf16* xb      = (__bf16*)alloc((size_t)M_ * IO_ * 2);
  __bf16* wbase_b = (__bf16*)alloc((size_t)D_ * IO_ * 2);
  __bf16* wA_b    = (__bf16*)alloc((size_t)RANK_ * IO_ * 2);
  __bf16* encBt_b = (__bf16*)alloc((size_t)ND_ * RANK_ * 2);
  __bf16* w1_b    = (__bf16*)alloc((size_t)HID_ * D_ * 2);
  __bf16* w2_b    = (__bf16*)alloc((size_t)D_ * HID_ * 2);
  __bf16* wqT     = (__bf16*)alloc((size_t)12 * DSLOT * D_ * 2);
  __bf16* wkT     = (__bf16*)alloc((size_t)12 * DSLOT * D_ * 2);
  __bf16* wvT     = (__bf16*)alloc((size_t)12 * DSLOT * D_ * 2);
  __bf16* wdown_b = (__bf16*)alloc((size_t)RANK_ * ND_ * 2);
  __bf16* wup_b   = (__bf16*)alloc((size_t)IO_ * RANK_ * 2);
  float*  basef   = (float*) alloc((size_t)M_ * D_ * 4);
  __bf16* hb      = (__bf16*)alloc((size_t)M_ * RANK_ * 2);
  float*  zf      = (float*) alloc((size_t)M_ * ND_ * 4);     // z -> z_head -> z_new
  __bf16* zmb     = (__bf16*)alloc((size_t)MN_ * D_ * 2);
  __bf16* ab      = (__bf16*)alloc((size_t)MN_ * HID_ * 2);
  __bf16* hinb    = (__bf16*)alloc((size_t)MN_ * D_ * 2);
  __bf16* qb      = (__bf16*)alloc((size_t)M_ * K_ * N_ * DSLOT * 2);
  __bf16* kb      = (__bf16*)alloc((size_t)M_ * K_ * N_ * DSLOT * 2);
  __bf16* vb      = (__bf16*)alloc((size_t)M_ * K_ * N_ * DSLOT * 2);
  __bf16* scores  = (__bf16*)alloc((size_t)48 * T_ * T_ * 2);
  __bf16* cb      = (__bf16*)alloc((size_t)M_ * ND_ * 2);
  __bf16* gb      = (__bf16*)alloc((size_t)M_ * RANK_ * 2);

  auto cdiv = [](long a, long b) { return (unsigned)((a + b - 1) / b); };

  // --- precision conversions (bf16 operand staging) ---
  k_cvt<<<cdiv((long)M_ * IO_, 256), 256, 0, stream>>>(x, xb, (long)M_ * IO_);
  k_cvt<<<cdiv((long)D_ * IO_, 256), 256, 0, stream>>>(encbasew, wbase_b, (long)D_ * IO_);
  k_cvt<<<cdiv((long)RANK_ * IO_, 256), 256, 0, stream>>>(encAw, wA_b, (long)RANK_ * IO_);
  k_tcvt<<<dim3(cdiv((long)RANK_ * ND_, 256), 1, 1), 256, 0, stream>>>(encB, encBt_b, RANK_, ND_);
  k_cvt<<<cdiv((long)HID_ * D_, 256), 256, 0, stream>>>(w1, w1_b, (long)HID_ * D_);
  k_cvt<<<cdiv((long)D_ * HID_, 256), 256, 0, stream>>>(w2, w2_b, (long)D_ * HID_);
  k_tcvt<<<dim3(cdiv((long)D_ * DSLOT, 256), 1, 12), 256, 0, stream>>>(Wq, wqT, D_, DSLOT);
  k_tcvt<<<dim3(cdiv((long)D_ * DSLOT, 256), 1, 12), 256, 0, stream>>>(Wk, wkT, D_, DSLOT);
  k_tcvt<<<dim3(cdiv((long)D_ * DSLOT, 256), 1, 12), 256, 0, stream>>>(Wv, wvT, D_, DSLOT);
  k_cvt<<<cdiv((long)RANK_ * ND_, 256), 256, 0, stream>>>(wdown, wdown_b, (long)RANK_ * ND_);
  k_cvt<<<cdiv((long)IO_ * RANK_, 256), 256, 0, stream>>>(wup, wup_b, (long)IO_ * RANK_);

  // --- encoder: base, h, z = base + beta*delta ---
  k_gemm_base<<<dim3(D_ / 128, M_ / 128), 256, 0, stream>>>(xb, wbase_b, basef);
  k_gemm_h<<<dim3(RANK_ / 128, M_ / 128), 256, 0, stream>>>(xb, wA_b, hb);
  k_gemm_z<<<dim3(ND_ / 128, M_ / 128), 256, 0, stream>>>(hb, encBt_b, basef, beta_p, zf);

  // --- mixer MLP ---
  k_ln<<<MN_, 256, 0, stream>>>(zf, lnm_g, lnm_b, zmb, D_);
  k_gemm_w1<<<dim3(HID_ / 128, MN_ / 128), 256, 0, stream>>>(zmb, w1_b, b1, ab);
  k_gemm_w2<<<dim3(D_ / 128, MN_ / 128), 256, 0, stream>>>(ab, w2_b, b2, zf);

  // --- attention ---
  k_ln<<<MN_, 256, 0, stream>>>(zf, lna_g, lna_b, hinb, D_);
  k_gemm_qkv<<<dim3(DSLOT / 128, M_ / 128, 36), 256, 0, stream>>>(
      hinb, wqT, wkT, wvT, t_p, qb, kb, vb);
  k_gemm_scores<<<dim3(T_ / 128, T_ / 128, 48), 256, 0, stream>>>(qb, kb, scores);
  k_softmax<<<48 * T_, 256, 0, stream>>>(scores);
  k_gemm_attout<<<dim3(DSLOT / 128, T_ / 128, 48), 256, 0, stream>>>(scores, vb, t_p, zf);

  // --- decoder ---
  k_ln<<<M_, 256, 0, stream>>>(zf, dln_g, dln_b, cb, ND_);
  k_gemm_down<<<dim3(RANK_ / 128, M_ / 128), 256, 0, stream>>>(cb, wdown_b, gb);
  k_gemm_up<<<dim3(IO_ / 128, M_ / 128), 256, 0, stream>>>(gb, wup_b, y);
}
